// LSTMLayer_19628000543504
// MI455X (gfx1250) — matrix operand order repaired and checked by reading, not compiled, not run on hardware
//
#include <hip/hip_runtime.h>

// ---------------------------------------------------------------------------
// 2-layer LSTM forward for MI455X (gfx1250, wave32, WMMA) — persistent kernel
// with LDS-resident weights.
//
// * Persistent: per layer-step the GEMM is [32x2048]@[2048x4096] ~ 537 MFLOP
//   (~1us of WMMA math) -> 1024 launches would cost more than the compute.
//   One kernel, 128 blocks (64 per layer, co-resident), 512-step loop inside,
//   agent-scope acquire/release atomics + s_sleep for cross-block sync.
//   Layer-0 free-runs ahead of layer-1 (its full f16 h trajectory lives in ws
//   and doubles as layer-1's GEMM input) -> true pipeline, ~512 serial phases.
//
// * LDS-resident weights (CDNA5: 320KB per WGP): each block's 64 gate-rows x
//   2048 K of f16 weights = 256KB are staged into LDS once, then all 512
//   steps read B fragments via ds_load_b128. Critical because occupancy is
//   1 wave/SIMD here (no other-wave latency hiding): L2 round-trips would sit
//   on the serial critical path. Weight L2 traffic: 17 GB -> 33.5 MB total.
//   Rows padded to 2056 f16 (stride = 4 dwords mod 64 banks, 16B aligned) to
//   avoid 16-way bank conflicts.
//
// * Each wave owns one gate, computes BOTH batch M-tiles -> every B fragment
//   feeds two WMMAs. Gates meet in LDS; sigmoid/tanh + c/h update fused.
// ---------------------------------------------------------------------------

typedef _Float16 f16;
typedef __attribute__((ext_vector_type(16))) _Float16 v16h;
typedef __attribute__((ext_vector_type(8)))  _Float16 v8h;
typedef __attribute__((ext_vector_type(8)))  float    v8f;

#define NL   2
#define NB   32
#define NT   512
#define ND   1024
#define NH   1024
#define G4H  4096            // 4*H
#define KCAT 2048            // D + H
#define BH   (NB * NH)       // 32768
#define LBH  (NL * BH)       // 65536
#define NBLK_PER_LAYER 64    // one per 16-wide H slab
#define WSTRIDE (KCAT + 8)   // padded LDS row stride (f16 units)

// ---------------- prep kernels (re-run every call: deterministic) -----------

__global__ void prep_weights(const float* __restrict__ Wih,
                             const float* __restrict__ Whh,
                             f16* __restrict__ wcat) {
    // wcat[l][n][k] : k<ND -> Wih[l][n][k] ; k>=ND -> Whh[l][n][k-ND]
    size_t idx = (size_t)blockIdx.x * blockDim.x + threadIdx.x;
    if (idx >= (size_t)NL * G4H * KCAT) return;
    int k = (int)(idx % KCAT);
    size_t ln = idx / KCAT;            // l*G4H + n
    float v = (k < ND) ? Wih[ln * ND + k] : Whh[ln * NH + (k - ND)];
    wcat[idx] = (f16)v;
}

__global__ void prep_x(const float* __restrict__ x, f16* __restrict__ xf) {
    size_t idx = (size_t)blockIdx.x * blockDim.x + threadIdx.x;
    if (idx >= (size_t)NB * NT * ND) return;
    xf[idx] = (f16)x[idx];
}

__global__ void init_state(const float* __restrict__ h0,
                           const float* __restrict__ c0,
                           float* __restrict__ cs,   // [L][B][H] fp32 cell state
                           f16* __restrict__ hL0,    // layer0 traj, slot 0 = h0
                           f16* __restrict__ hp1,    // layer1 ping-pong, slot 0
                           int* __restrict__ cnt) {  // [2][T] sync counters
    size_t idx = (size_t)blockIdx.x * blockDim.x + threadIdx.x;
    if (idx < (size_t)2 * NT) cnt[idx] = 0;
    if (idx >= (size_t)LBH) return;
    cs[idx] = c0[idx];
    float h = h0[idx];
    if (idx < BH) hL0[idx]      = (f16)h;   // layer 0 initial h
    else          hp1[idx - BH] = (f16)h;   // layer 1 initial h
}

// ---------------- persistent recurrence kernel ------------------------------

__device__ __forceinline__ float sigmoidf_(float v) {
    return 1.0f / (1.0f + __expf(-v));
}

__device__ __forceinline__ v16h load_a(const f16* __restrict__ row, int kb, int hi) {
    // 16-bit A fragment, 16x32: lane half `hi` holds K runs [kb+8hi, +8) and
    // [kb+16+8hi, +8) -> two aligned 16B loads.
    const v8h a0 = *(const v8h*)(row + kb + hi * 8);
    const v8h a1 = *(const v8h*)(row + kb + 16 + hi * 8);
    v16h a;
#pragma unroll
    for (int e = 0; e < 8; ++e) { a[e] = a0[e]; a[e + 8] = a1[e]; }
    return a;
}

__device__ __forceinline__ v8f wmma_(v16h a, v16h b, v8f c) {
    return __builtin_amdgcn_wmma_f32_16x16x32_f16(
               false, a, false, b, (short)0, c, false, false);
}

__global__ void __launch_bounds__(128)
lstm_persistent(const f16* __restrict__ xf,     // [B][T][D] f16
                const f16* __restrict__ wcat,   // [L][4096][2048] f16
                const float* __restrict__ bias, // [L][4096]
                f16* __restrict__ hL0,          // [T+1][B][H] layer0 h traj
                f16* __restrict__ hp1,          // [2][B][H] layer1 h ping-pong
                float* __restrict__ cs,         // [L][B][H] cell state
                float* __restrict__ out,        // ys + hT + cT
                int* __restrict__ cnt)          // [2][T]
{
    // CDNA5 WGP LDS: 320KB. Weights slab (256KB, padded) + gate exchange.
    __shared__ __align__(16) f16 wlds[64 * WSTRIDE];   // 263,168 B
    __shared__ float lds_g[4][32][17];                 //   8,704 B

    const int tid   = threadIdx.x;
    const int lane  = tid & 31;
    const int gate  = tid >> 5;          // 4 waves: 0=i 1=f 2=g 3=o
    const int hi    = lane >> 4;
    const int lq    = lane & 15;
    const int layer = (int)(blockIdx.x >> 6);   // 0..1
    const int slab  = (int)(blockIdx.x & 63);   // 16-wide H slab

    const int ncol = gate * NH + slab * 16 + lq;            // this lane's gate row
    const float bv = bias[layer * G4H + ncol];
    float* cSt   = cs + (size_t)layer * BH;
    float* tailH = out + (size_t)NB * NT * NH + (size_t)layer * BH;
    float* tailC = tailH + (size_t)LBH;

    int* myCnt = cnt + layer * NT;
    int* l0Cnt = cnt;                    // layer-0 progress

    // ---- stage this block's 64 weight rows into LDS (once for all 512 t) ---
    {
        const f16* src0 = wcat + (size_t)layer * G4H * KCAT;
        for (int lr = 0; lr < 64; ++lr) {
            const int n = (lr >> 4) * NH + slab * 16 + (lr & 15);
            const f16* src = src0 + (size_t)n * KCAT;
            f16* dst = wlds + (size_t)lr * WSTRIDE;
            for (int kk = tid * 8; kk < KCAT; kk += 128 * 8)
                *(v8h*)(dst + kk) = *(const v8h*)(src + kk);
        }
        __syncthreads();
    }
    const int lrow = gate * 16 + lq;                    // lane's LDS weight row
    const f16* wl  = wlds + (size_t)lrow * WSTRIDE;     // x part (k 0..1023)
    const f16* wlH = wl + ND;                           // h part

    for (int t = 0; t < NT; ++t) {
        // ---- cross-block dependency sync (thread 0 spins, block follows) ----
        if (tid == 0) {
            if (layer == 0) {
                if (t > 0)
                    while (__hip_atomic_load(&l0Cnt[t - 1], __ATOMIC_ACQUIRE,
                                             __HIP_MEMORY_SCOPE_AGENT) < NBLK_PER_LAYER)
                        __builtin_amdgcn_s_sleep(2);
            } else {
                while (__hip_atomic_load(&l0Cnt[t], __ATOMIC_ACQUIRE,
                                         __HIP_MEMORY_SCOPE_AGENT) < NBLK_PER_LAYER)
                    __builtin_amdgcn_s_sleep(2);
                if (t > 0)
                    while (__hip_atomic_load(&myCnt[t - 1], __ATOMIC_ACQUIRE,
                                             __HIP_MEMORY_SCOPE_AGENT) < NBLK_PER_LAYER)
                        __builtin_amdgcn_s_sleep(2);
            }
        }
        __syncthreads();

        // ---- per-step operand bases ----
        const f16* Abase;  size_t strideA;  const f16* Hbase;  f16* hOut;
        if (layer == 0) {
            Abase   = xf + (size_t)t * ND;       strideA = (size_t)NT * ND;
            Hbase   = hL0 + (size_t)t * BH;      // h_{t-1}
            hOut    = hL0 + (size_t)(t + 1) * BH;
        } else {
            Abase   = hL0 + (size_t)(t + 1) * BH; strideA = NH;  // layer0 h_t
            Hbase   = hp1 + (size_t)(t & 1) * BH;
            hOut    = hp1 + (size_t)((t + 1) & 1) * BH;
        }
        const f16* ar0 = Abase + (size_t)lq * strideA;
        const f16* ar1 = Abase + (size_t)(16 + lq) * strideA;
        const f16* hr0 = Hbase + (size_t)lq * NH;
        const f16* hr1 = Hbase + (size_t)(16 + lq) * NH;

        // ---- gates GEMM: K = [x_t ; h_{t-1}] = 2048, both M tiles per wave,
        //      B fragments from LDS (ds_load_b128), A from global f16 --------
        v8f acc0 = {}, acc1 = {};
#pragma unroll 4
        for (int kc = 0; kc < ND / 32; ++kc) {
            const int kb = kc * 32;
            const v16h b = load_a(wl, kb, hi);
            acc0 = wmma_(load_a(ar0, kb, hi), b, acc0);   // B frag reused:
            acc1 = wmma_(load_a(ar1, kb, hi), b, acc1);   // 2 WMMAs per read
        }
#pragma unroll 4
        for (int kc = 0; kc < NH / 32; ++kc) {
            const int kb = kc * 32;
            const v16h b = load_a(wlH, kb, hi);
            acc0 = wmma_(load_a(hr0, kb, hi), b, acc0);
            acc1 = wmma_(load_a(hr1, kb, hi), b, acc1);
        }
#pragma unroll
        for (int r = 0; r < 8; ++r) { acc0[r] += bv; acc1[r] += bv; }

        // ---- gate tiles -> LDS (C layout: VGPR r -> M = r + 8*hi) ----
#pragma unroll
        for (int r = 0; r < 8; ++r) {
            lds_g[gate][hi * 8 + r][lq]      = acc0[r];
            lds_g[gate][16 + hi * 8 + r][lq] = acc1[r];
        }
        __syncthreads();

        // ---- fused pointwise: 32 batches x 16 cols, 4 per thread ----
#pragma unroll
        for (int it = 0; it < 4; ++it) {
            const int idx  = tid + it * 128;
            const int m    = idx >> 4;
            const int hl   = idx & 15;
            const int hcol = slab * 16 + hl;

            const float ig = sigmoidf_(lds_g[0][m][hl]);
            const float fg = sigmoidf_(lds_g[1][m][hl]);
            const float gg = tanhf    (lds_g[2][m][hl]);
            const float og = sigmoidf_(lds_g[3][m][hl]);

            const size_t si = (size_t)m * NH + hcol;
            const float ct = fg * cSt[si] + ig * gg;   // c owned by this thread
            const float ht = og * tanhf(ct);

            cSt[si]  = ct;
            hOut[si] = (f16)ht;
            if (layer == 1)
                out[(size_t)m * (NT * NH) + (size_t)t * NH + hcol] = ht;
            if (t == NT - 1) { tailH[si] = ht; tailC[si] = ct; }
        }

        // ---- publish h_t: flush this block's writes, then release-signal ----
        __threadfence();
        __syncthreads();
        if (tid == 0)
            __hip_atomic_fetch_add(&myCnt[t], 1, __ATOMIC_RELEASE,
                                   __HIP_MEMORY_SCOPE_AGENT);
    }
}

// ---------------- host-side orchestration -----------------------------------

extern "C" void kernel_launch(void* const* d_in, const int* in_sizes, int n_in,
                              void* d_out, int out_size, void* d_ws, size_t ws_size,
                              hipStream_t stream) {
    const float* x   = (const float*)d_in[0];
    const float* h0  = (const float*)d_in[1];
    const float* c0  = (const float*)d_in[2];
    const float* Wih = (const float*)d_in[3];
    const float* bih = (const float*)d_in[4];
    const float* Whh = (const float*)d_in[5];
    float* out = (float*)d_out;

    // workspace layout (all chunks 256B-aligned)
    char* ws = (char*)d_ws;
    const size_t wcat_bytes = (size_t)NL * G4H * KCAT * sizeof(f16);    // 33.55 MB
    const size_t xf_bytes   = (size_t)NB * NT * ND * sizeof(f16);       // 33.55 MB
    const size_t hL0_bytes  = (size_t)(NT + 1) * BH * sizeof(f16);      // 33.62 MB
    const size_t hp1_bytes  = (size_t)2 * BH * sizeof(f16);
    const size_t cs_bytes   = (size_t)LBH * sizeof(float);

    f16*   wcat = (f16*)ws;
    f16*   xf   = (f16*)(ws + wcat_bytes);
    f16*   hL0  = (f16*)(ws + wcat_bytes + xf_bytes);
    f16*   hp1  = (f16*)(ws + wcat_bytes + xf_bytes + hL0_bytes);
    float* cs   = (float*)(ws + wcat_bytes + xf_bytes + hL0_bytes + hp1_bytes);
    int*   cnt  = (int*)(ws + wcat_bytes + xf_bytes + hL0_bytes + hp1_bytes + cs_bytes);

    // one-time prep (weights/x -> f16, states + counters reset)
    const size_t nW = (size_t)NL * G4H * KCAT;
    prep_weights<<<(unsigned)((nW + 255) / 256), 256, 0, stream>>>(Wih, Whh, wcat);
    const size_t nX = (size_t)NB * NT * ND;
    prep_x<<<(unsigned)((nX + 255) / 256), 256, 0, stream>>>(x, xf);
    init_state<<<(LBH + 255) / 256, 256, 0, stream>>>(h0, c0, cs, hL0, hp1, cnt);

    // the whole 512-step, 2-layer recurrence in one persistent kernel:
    // 128 blocks (64 per layer) x 128 threads — co-resident; each block's
    // weight slab lives in its WGP's 320KB LDS for the entire recurrence.
    lstm_persistent<<<2 * NBLK_PER_LAYER, 128, 0, stream>>>(
        xf, wcat, bih, hL0, hp1, cs, out, cnt);
}
